// EncoderDistillationLoss_44263932953089
// MI455X (gfx1250) — compile-verified, matrix operand order checked by reading, not run on hardware
//
#include <hip/hip_runtime.h>
#include <hip/hip_bf16.h>
#include <math.h>

// EncoderDistillationLoss — MI455X (gfx1250, wave32, WMMA bf16 16x16x32)
//
// Forward-only: STE means softmaxes vanish; loss = vq MSE + three per-row
// scans over the (N,K) distance field, fused into the N x K x C GEMM so the
// 196MB distance matrix is never written. Both the codebook B-strips and the
// gumbel tiles are double-buffered in LDS via global_load_async_to_lds_b128
// (ASYNCcnt-gated), so the WMMA chain runs ds_load_b128 -> v_wmma with no
// exposed L2 latency.

typedef __attribute__((ext_vector_type(16))) __bf16 v16bf;
typedef __attribute__((ext_vector_type(8)))  __bf16 v8bf;
typedef __attribute__((ext_vector_type(8)))  float  v8f;

static constexpr int Bn = 8;
static constexpr int Cn = 512;
static constexpr int Tn = 1500;
static constexpr int Kn = 4096;
static constexpr int Nn = Bn * Tn;      // 12000 rows
static constexpr int KTILES = Kn / 16;  // 256
static constexpr int BPAD = 520;        // 512 + 8 bf16 pad (1040B = 65*16)

struct RowPart {
    float gmax; int gidx;
    float dmin; int didx;
    float dwrong;
    float dtc;
};

// 16B async copy global -> LDS (per-lane), tracked by ASYNCcnt.
__device__ __forceinline__ void async_copy16(unsigned lds_byte_addr, const void* gsrc) {
    asm volatile("global_load_async_to_lds_b128 %0, %1, off"
                 :: "v"(lds_byte_addr),
                    "v"((unsigned long long)(uintptr_t)gsrc)
                 : "memory");
}
__device__ __forceinline__ unsigned lds_offset_of(const void* p) {
    // generic LDS pointer = aperture(high word) | byte-offset(low word)
    return (unsigned)(uintptr_t)p;
}

// ---------------------------------------------------------------- init ----
__global__ void edl_init_kernel(float* out) {
    if (threadIdx.x == 0) out[0] = 0.0f;
}

// ------------------------------------------------- codebook prep (bf16) ----
__global__ __launch_bounds__(256) void edl_prep_kernel(
    const float* __restrict__ cb, __bf16* __restrict__ cbh,
    float* __restrict__ cnorm)
{
    const int k = blockIdx.x;
    const float* row = cb + (size_t)k * Cn;
    __bf16* orow = cbh + (size_t)k * Cn;
    float s = 0.0f;
    for (int c = threadIdx.x; c < Cn; c += 256) {
        float v = row[c];
        orow[c] = (__bf16)v;
        s += v * v;
    }
    __shared__ float red[256];
    red[threadIdx.x] = s;
    __syncthreads();
    for (int off = 128; off > 0; off >>= 1) {
        if ((int)threadIdx.x < off) red[threadIdx.x] += red[threadIdx.x + off];
        __syncthreads();
    }
    if (threadIdx.x == 0) cnorm[k] = red[0];
}

// ------------------------------------------------------------ vq loss ----
__global__ __launch_bounds__(256) void edl_vq_kernel(
    const float* __restrict__ sf, const float* __restrict__ cb,
    const int* __restrict__ tcodes, float* __restrict__ out)
{
    const int tpb = (Tn + 15) / 16;            // 94 t-tiles per batch
    const int bb  = blockIdx.x / tpb;
    const int t0  = (blockIdx.x % tpb) * 16;
    const int tl  = threadIdx.x & 15;
    const int cl  = threadIdx.x >> 4;
    const int t   = t0 + tl;

    float s = 0.0f;
    if (t < Tn) {
        const int n = bb * Tn + t;
        const int code = tcodes[n];
        const float* crow = cb + (size_t)code * Cn;
        const float* sfp  = sf + (size_t)bb * (size_t)(Cn * Tn) + t;
        for (int c0 = 0; c0 < Cn; c0 += 16) {
            const int c = c0 + cl;
            float f = sfp[(size_t)c * Tn];
            float d = f - crow[c];
            s += d * d;
        }
    }
    __shared__ float red[256];
    red[threadIdx.x] = s;
    __syncthreads();
    for (int off = 128; off > 0; off >>= 1) {
        if ((int)threadIdx.x < off) red[threadIdx.x] += red[threadIdx.x + off];
        __syncthreads();
    }
    if (threadIdx.x == 0)
        atomicAdd(out, red[0] * (1.0f / ((float)Nn * (float)Cn)));
}

// --------------------------------------------------------- main kernel ----
__global__ __launch_bounds__(256) void edl_main_kernel(
    const float*  __restrict__ sf,     // (B, C, T) f32
    const __bf16* __restrict__ cbh,    // (K, C) bf16
    const float*  __restrict__ cnorm,  // (K,)  ||c||^2
    const float*  __restrict__ gum,    // (N, K) f32
    const float*  __restrict__ dm,     // (K, K) f32
    const int*    __restrict__ tcodes, // (N,)
    float*        __restrict__ out)
{
    const int wave = threadIdx.x >> 5;   // 0..7
    const int lane = threadIdx.x & 31;
    const int rt   = wave & 1;           // row-tile within block
    const int ks   = wave >> 1;          // k-set 0..3 (shared by wave pair)
    const int half = lane >> 4;
    const int r    = lane & 15;

    const int n_block0 = blockIdx.x * 32;
    const int n_base   = n_block0 + rt * 16;
    const int nrow     = n_base + r;
    const int bb       = nrow / Tn;
    const int tt       = nrow % Tn;

    // ---- preload A fragments (WMMA 16-bit A layout) + f32 row norm ----
    const float* sfb = sf + (size_t)bb * (size_t)(Cn * Tn) + tt;
    v16bf afrag[16];
    float fn = 0.0f;
#pragma unroll
    for (int cc = 0; cc < 16; ++cc) {
        const int c0 = cc * 32;
#pragma unroll
        for (int e = 0; e < 8; ++e) {
            float v0 = sfb[(size_t)(c0 + half * 8 + e) * Tn];
            float v1 = sfb[(size_t)(c0 + 16 + half * 8 + e) * Tn];
            fn += v0 * v0 + v1 * v1;
            afrag[cc][e]     = (__bf16)v0;
            afrag[cc][8 + e] = (__bf16)v1;
        }
    }
    fn += __shfl_xor(fn, 16, 32);
    const int tcode = tcodes[nrow];

    // running per-row state (replicated in lanes r and r+16)
    float gmax = -INFINITY; int gidx = 0;
    float dmin =  INFINITY; int didx = 0;
    float dwrong = INFINITY;
    float dtc = -1.0f;

    // LDS: B strips (wave-pair shared, dbl-buffered), gumbel tiles, scratch
    __shared__ __align__(16) __bf16 Bstrip[4][2][16][BPAD];   // 130KB
    __shared__ __align__(16) float  gtile[8][2][16][20];      // 20KB, 80B rows
    union Tail {
        float dtile[8][16][17];                               // dot exchange
        struct { RowPart part[32][4]; float contrib[32]; } fin;
    };
    __shared__ Tail tail;

    // -------- gumbel async staging: lane L copies row L>>1, 16B segs ------
    const int grow_l = lane >> 1;
    const int gseg   = lane & 1;
    const char* gsrc_row = (const char*)(gum + (size_t)(n_base + grow_l) * Kn);
    const unsigned gt0 = lds_offset_of(&gtile[wave][0][grow_l][0]);
    const unsigned gt1 = lds_offset_of(&gtile[wave][1][grow_l][0]);
    auto issue_gumbel = [&](int k0, int buf) {
        const unsigned dst = (buf ? gt1 : gt0);
        const char* src = gsrc_row + (size_t)k0 * 4;
        async_copy16(dst + (unsigned)(gseg * 16),       src + gseg * 16);
        async_copy16(dst + (unsigned)((gseg + 2) * 16), src + (gseg + 2) * 16);
    };

    // -------- B async staging: pair splits 16 cols; lane -> col, 16B seg --
    const int bcol = rt * 8 + (lane >> 2);       // this lane's codebook col
    const int bseg = (lane & 3) * 16;            // byte seg within 64B group
    const unsigned bs0 = lds_offset_of(&Bstrip[ks][0][bcol][0]) + (unsigned)bseg;
    const unsigned bs1 = lds_offset_of(&Bstrip[ks][1][bcol][0]) + (unsigned)bseg;
    auto issue_B = [&](int k0, int buf) {
        const unsigned dst = (buf ? bs1 : bs0);
        const char* src = (const char*)cbh + (size_t)(k0 + bcol) * (Cn * 2) + bseg;
#pragma unroll
        for (int j = 0; j < 16; ++j)
            async_copy16(dst + (unsigned)(j * 64), src + j * 64);
    };

    int cur = 0;
    issue_B(ks * 16, 0);          // prologue: 16 async ops
    issue_gumbel(ks * 16, 0);     // + 2 async ops

    for (int kt = ks; kt < KTILES; kt += 4) {
        const int k0 = kt * 16;
        const bool has_next = (kt + 4 < KTILES);

        if (has_next) {
            issue_B((kt + 4) * 16, cur ^ 1);
            issue_gumbel((kt + 4) * 16, cur ^ 1);
        }
        // current buffer's 18 copies landed (in-order completion); only the
        // 18 just issued may remain outstanding.
        if (has_next) asm volatile("s_wait_asynccnt 18" ::: "memory");
        else          asm volatile("s_wait_asynccnt 0"  ::: "memory");
        __syncthreads();   // publish async LDS writes to the wave pair

        // ---- GEMM tile: ds_load_b128 -> v_wmma, 16 chunks over C ----
        v8f acc = {0.f, 0.f, 0.f, 0.f, 0.f, 0.f, 0.f, 0.f};
#pragma unroll
        for (int cc = 0; cc < 16; ++cc) {
            const __bf16* bp = &Bstrip[ks][cur][r][cc * 32 + half * 8];
            v8bf lo = *(const v8bf*)(bp);
            v8bf hi = *(const v8bf*)(bp + 16);
            v16bf bf;
#pragma unroll
            for (int e = 0; e < 8; ++e) { bf[e] = lo[e]; bf[8 + e] = hi[e]; }
            acc = __builtin_amdgcn_wmma_f32_16x16x32_bf16(
                false, afrag[cc], false, bf, (short)0, acc, false, false);
        }

        // scatter dots (m, col) -> LDS (same-wave LDS is in-order)
#pragma unroll
        for (int e = 0; e < 8; ++e) {
            const int m = half ? (8 + e) : e;
            tail.dtile[wave][m][r] = acc[e];
        }
        asm volatile("s_wait_dscnt 0" ::: "memory");

        // reassigned view: lane scans row r, cols k0+half*8 .. +7
        const int kb = k0 + half * 8;
        float lg = -INFINITY; int lgi = 0;
        float ld =  INFINITY; int ldi = 0;
        float lw =  INFINITY;
        float lt = -1.0f;
#pragma unroll
        for (int e = 0; e < 8; ++e) {
            const int k = kb + e;
            const float dot = tail.dtile[wave][r][half * 8 + e];
            const float sq  = fn + cnorm[k] - 2.0f * dot;
            const float d   = sqrtf(fmaxf(sq, 1e-12f));
            const float g   = gtile[wave][cur][r][half * 8 + e] - d;
            if (g > lg) { lg = g; lgi = k; }
            if (d < ld) { ld = d; ldi = k; }
            if (k == tcode) lt = d; else lw = fminf(lw, d);
        }
        // merge lane pair (r <-> r+16)
        {
            float og = __shfl_xor(lg, 16, 32); int ogi = __shfl_xor(lgi, 16, 32);
            if (og > lg || (og == lg && ogi < lgi)) { lg = og; lgi = ogi; }
            float od = __shfl_xor(ld, 16, 32); int odi = __shfl_xor(ldi, 16, 32);
            if (od < ld || (od == ld && odi < ldi)) { ld = od; ldi = odi; }
            lw = fminf(lw, __shfl_xor(lw, 16, 32));
            lt = fmaxf(lt, __shfl_xor(lt, 16, 32));
        }
        if (lg > gmax) { gmax = lg; gidx = lgi; }
        if (ld < dmin) { dmin = ld; didx = ldi; }
        dwrong = fminf(dwrong, lw);
        dtc    = fmaxf(dtc, lt);

        __syncthreads();   // all reads of buf `cur` done before it is refilled
        cur ^= 1;
    }

    // ---- cross-k-set merge + per-row loss terms (dtile storage reused) ----
    if (lane < 16) {
        RowPart p{gmax, gidx, dmin, didx, dwrong, dtc};
        tail.fin.part[rt * 16 + r][ks] = p;
    }
    __syncthreads();
    if (ks == 0 && lane < 16) {
        const int row = rt * 16 + r;
        RowPart p = tail.fin.part[row][0];
        for (int j = 1; j < 4; ++j) {
            RowPart q = tail.fin.part[row][j];
            if (q.gmax > p.gmax || (q.gmax == p.gmax && q.gidx < p.gidx)) { p.gmax = q.gmax; p.gidx = q.gidx; }
            if (q.dmin < p.dmin || (q.dmin == p.dmin && q.didx < p.didx)) { p.dmin = q.dmin; p.didx = q.didx; }
            p.dwrong = fminf(p.dwrong, q.dwrong);
            p.dtc    = fmaxf(p.dtc, q.dtc);
        }
        const int tcr = tcodes[n_block0 + row];
        const float margin = fmaxf(p.dtc - p.dwrong + 1.0f, 0.0f);
        const float gml  = dm[(size_t)tcr * Kn + p.gidx];
        const float stel = dm[(size_t)tcr * Kn + p.didx];
        tail.fin.contrib[row] = margin + gml + stel;
    }
    __syncthreads();
    if (threadIdx.x == 0) {
        float s = 0.0f;
        for (int i = 0; i < 32; ++i) s += tail.fin.contrib[i];
        atomicAdd(out, s * (1.0f / (float)Nn));
    }
}

// ------------------------------------------------------------- launch ----
extern "C" void kernel_launch(void* const* d_in, const int* in_sizes, int n_in,
                              void* d_out, int out_size, void* d_ws, size_t ws_size,
                              hipStream_t stream) {
    (void)in_sizes; (void)n_in; (void)out_size; (void)ws_size;
    const float* sf     = (const float*)d_in[0];   // (B, C, T)
    const float* cb     = (const float*)d_in[1];   // (K, C)
    const float* dm     = (const float*)d_in[2];   // (K, K)
    const float* gum    = (const float*)d_in[3];   // (N, K)
    const int*   tcodes = (const int*)d_in[4];     // (NQ, B, T) -> use [0]
    float* out = (float*)d_out;

    // workspace: bf16 codebook (K*C*2 = 4MB) then f32 norms (K*4)
    __bf16* cbh   = (__bf16*)d_ws;
    float*  cnorm = (float*)((char*)d_ws + (size_t)Kn * Cn * sizeof(__bf16));

    edl_init_kernel<<<1, 32, 0, stream>>>(out);
    edl_prep_kernel<<<Kn, 256, 0, stream>>>(cb, cbh, cnorm);
    edl_vq_kernel<<<Bn * ((Tn + 15) / 16), 256, 0, stream>>>(sf, cb, tcodes, out);
    edl_main_kernel<<<Nn / 32, 256, 0, stream>>>(sf, cbh, cnorm, gum, dm, tcodes, out);
}
